// RelPartialLearnableMultiHeadAttn_59124519796958
// MI455X (gfx1250) — matrix operand order repaired and checked by reading, not compiled, not run on hardware
//
#include <hip/hip_runtime.h>
#include <hip/hip_bf16.h>

typedef _Float16 h16 __attribute__((ext_vector_type(16)));
typedef float    f8  __attribute__((ext_vector_type(8)));

constexpr int QL = 2048, BS = 2, NH = 16, DH = 64, DM = 1024;
constexpr int TOK = QL * BS;                 // 4096 tokens, token t = i*BS + b
constexpr float SCL = 0.125f;                // 1/sqrt(64)

// ---------------- WMMA fragment helpers (16x16x32 f16, wave32) ----------------
// A (16xK=32): lane L holds row M=L&15; K chunk base 8*(L>>4) and 16+8*(L>>4).
__device__ __forceinline__ h16 ldgA(const _Float16* base, int ld, int r0, int c0) {
    int lane = threadIdx.x & 31, g = lane >> 4;
    const _Float16* p = base + (size_t)(r0 + (lane & 15)) * ld + c0 + 8 * g;
    h16 a;
    ((uint4*)&a)[0] = *(const uint4*)(p);
    ((uint4*)&a)[1] = *(const uint4*)(p + 16);
    return a;
}
// B (K=32 x 16) with B(k,n) = Y[n0+n][c0+k] (Y row-major): lane holds col n=L&15,
// K chunk base 8*(L>>4) and 16+8*(L>>4) (same as A).
__device__ __forceinline__ h16 ldgB(const _Float16* base, int ld, int n0, int c0) {
    int lane = threadIdx.x & 31, g = lane >> 4;
    const _Float16* p = base + (size_t)(n0 + (lane & 15)) * ld + c0 + 8 * g;
    h16 b;
    ((uint4*)&b)[0] = *(const uint4*)(p);
    ((uint4*)&b)[1] = *(const uint4*)(p + 16);
    return b;
}
__device__ __forceinline__ h16 ldgB_clamp(const _Float16* base, int ld, int n0, int c0, int maxrow) {
    int lane = threadIdx.x & 31, g = lane >> 4;
    int r = n0 + (lane & 15); if (r > maxrow) r = maxrow;
    const _Float16* p = base + (size_t)r * ld + c0 + 8 * g;
    h16 b;
    ((uint4*)&b)[0] = *(const uint4*)(p);
    ((uint4*)&b)[1] = *(const uint4*)(p + 16);
    return b;
}
__device__ __forceinline__ f8 wmma16(h16 a, h16 b, f8 c) {
    return __builtin_amdgcn_wmma_f32_16x16x32_f16(false, a, false, b, (short)0, c, false, false);
}

// ---------------- elementwise f32 -> f16 ----------------
__global__ void k_cvt(const float* __restrict__ in, _Float16* __restrict__ out, int n) {
    int i = blockIdx.x * blockDim.x + threadIdx.x;
    int st = gridDim.x * blockDim.x;
    for (; i < n; i += st) out[i] = (_Float16)in[i];
}

// ---------------- QKV projection: [4096 tok] x [3072 out], K=1024 ----------------
// Writes Qw=(q+rwb)*SCL, Qr=(q+rrb)*SCL, K [b,h,j,d], V transposed [b,h,d,j] (all f16).
__global__ __launch_bounds__(128) void k_qkv(const _Float16* __restrict__ A,
                                             const _Float16* __restrict__ W,
                                             const float* __restrict__ rwb,
                                             const float* __restrict__ rrb,
                                             _Float16* __restrict__ Qw, _Float16* __restrict__ Qr,
                                             _Float16* __restrict__ Kf, _Float16* __restrict__ Vt) {
    int wid = blockIdx.x * 4 + (threadIdx.x >> 5);
    int m0 = (wid & 255) * 16, n0 = (wid >> 8) * 16;
    f8 acc = {0,0,0,0,0,0,0,0};
    for (int kk = 0; kk < DM; kk += 32)
        acc = wmma16(ldgA(A, DM, m0, kk), ldgB(W, DM, n0, kk), acc);
    int lane = threadIdx.x & 31, g = lane >> 4, nn = lane & 15;
    int h = n0 + nn, sec = h >> 10, hh = h & 1023, head = hh >> 6, d = hh & 63;
#pragma unroll
    for (int v = 0; v < 8; ++v) {
        int tok = m0 + v + 8 * g, i = tok >> 1, bb = tok & 1;
        float val = acc[v];
        if (sec == 0) {
            size_t idx = ((size_t)(bb * NH + head) * QL + i) * DH + d;
            Qw[idx] = (_Float16)((val + rwb[head * DH + d]) * SCL);
            Qr[idx] = (_Float16)((val + rrb[head * DH + d]) * SCL);
        } else if (sec == 1) {
            Kf[((size_t)(bb * NH + head) * QL + i) * DH + d] = (_Float16)val;
        } else {
            Vt[((size_t)(bb * NH + head) * DH + d) * QL + i] = (_Float16)val;
        }
    }
}

// ---------------- R projection: [2048] x [1024], K=1024 -> RK[h,m,d] f16 ----------------
__global__ __launch_bounds__(128) void k_rproj(const _Float16* __restrict__ R,
                                               const _Float16* __restrict__ Wr,
                                               _Float16* __restrict__ RK) {
    int wid = blockIdx.x * 4 + (threadIdx.x >> 5);
    int m0 = (wid & 127) * 16, n0 = (wid >> 7) * 16;
    f8 acc = {0,0,0,0,0,0,0,0};
    for (int kk = 0; kk < DM; kk += 32)
        acc = wmma16(ldgA(R, DM, m0, kk), ldgB(Wr, DM, n0, kk), acc);
    int lane = threadIdx.x & 31, g = lane >> 4, nn = lane & 15;
    int head = (n0 + nn) >> 6, d = (n0 + nn) & 63;
#pragma unroll
    for (int v = 0; v < 8; ++v)
        RK[((size_t)head * QL + (m0 + v + 8 * g)) * DH + d] = (_Float16)acc[v];
}

// ---------------- flash attention with Transformer-XL relative shift ----------------
// One wave per (b, head, 16-row q-tile). Streams 32 k-cols per step.
__global__ __launch_bounds__(128) void k_attn(const _Float16* __restrict__ Qw,
                                              const _Float16* __restrict__ Qr,
                                              const _Float16* __restrict__ Kf,
                                              const _Float16* __restrict__ Vt,
                                              const _Float16* __restrict__ RK,
                                              _Float16* __restrict__ AV) {
    __shared__ __align__(16) float    strip[4][16 * 48];
    __shared__ __align__(16) _Float16 pbuf[4][16 * 32];
    int warp = threadIdx.x >> 5;
    int lane = threadIdx.x & 31, g = lane >> 4, nn = lane & 15;
    int wid = blockIdx.x * 4 + warp;
    int bh = wid >> 7;                 // b*16 + head
    int i0 = (wid & 127) << 4;
    int head = bh & 15;
    const _Float16* Qwp = Qw + (size_t)bh * QL * DH;
    const _Float16* Qrp = Qr + (size_t)bh * QL * DH;
    const _Float16* Kp  = Kf + (size_t)bh * QL * DH;
    const _Float16* Vp  = Vt + (size_t)bh * DH * QL;
    const _Float16* RKp = RK + (size_t)head * QL * DH;
    float*    strp = strip[warp];
    _Float16* pb   = pbuf[warp];

    // loop-invariant Q fragments (d = 0..31, 32..63)
    h16 aw0 = ldgA(Qwp, DH, i0, 0), aw1 = ldgA(Qwp, DH, i0, 32);
    h16 ar0 = ldgA(Qrp, DH, i0, 0), ar1 = ldgA(Qrp, DH, i0, 32);

    f8 o0 = {0,0,0,0,0,0,0,0}, o1 = o0, o2 = o0, o3 = o0;
    float mr[8], lr[8];
#pragma unroll
    for (int v = 0; v < 8; ++v) { mr[v] = -1e30f; lr[v] = 0.f; }

    int jend = i0 + 15;
    for (int j0 = 0; j0 <= jend; j0 += 32) {
        if (j0 + 32 <= jend) {
            __builtin_prefetch(Kp + (size_t)(j0 + 32) * DH, 0, 0);
            __builtin_prefetch(Vp + (size_t)j0 + 32, 0, 0);
        }
        // AC = Qw_tile x K_tile^T   (two 16-col sub-tiles)
        f8 s0 = {0,0,0,0,0,0,0,0}, s1 = s0;
        s0 = wmma16(aw0, ldgB(Kp, DH, j0,      0), s0);
        s0 = wmma16(aw1, ldgB(Kp, DH, j0,     32), s0);
        s1 = wmma16(aw0, ldgB(Kp, DH, j0 + 16, 0), s1);
        s1 = wmma16(aw1, ldgB(Kp, DH, j0 + 16,32), s1);
        // BD strip = Qr_tile x RK[mstart .. mstart+47]^T (rel-shift band)
        int mstart = QL - 16 - i0 + j0;
        f8 b0 = {0,0,0,0,0,0,0,0}, b1 = b0, b2 = b0;
        b0 = wmma16(ar0, ldgB_clamp(RKp, DH, mstart,      0, QL - 1), b0);
        b0 = wmma16(ar1, ldgB_clamp(RKp, DH, mstart,     32, QL - 1), b0);
        b1 = wmma16(ar0, ldgB_clamp(RKp, DH, mstart + 16, 0, QL - 1), b1);
        b1 = wmma16(ar1, ldgB_clamp(RKp, DH, mstart + 16,32, QL - 1), b1);
        b2 = wmma16(ar0, ldgB_clamp(RKp, DH, mstart + 32, 0, QL - 1), b2);
        b2 = wmma16(ar1, ldgB_clamp(RKp, DH, mstart + 32,32, QL - 1), b2);
        // strip -> LDS (per-wave region), then diagonal pick
        asm volatile("s_wait_dscnt 0" ::: "memory");
#pragma unroll
        for (int v = 0; v < 8; ++v) {
            int M = v + 8 * g;
            strp[M * 48 + nn]      = b0[v];
            strp[M * 48 + 16 + nn] = b1[v];
            strp[M * 48 + 32 + nn] = b2[v];
        }
        asm volatile("s_wait_dscnt 0" ::: "memory");
        float p0[8], p1[8];
#pragma unroll
        for (int v = 0; v < 8; ++v) {
            int M = v + 8 * g, irow = i0 + M;
            float x0 = s0[v] + strp[M * 48 + 15 - M + nn];       // m = mstart + 15-M+nn
            float x1 = s1[v] + strp[M * 48 + 31 - M + nn];
            if (j0 + nn > irow)      x0 = -1e30f;                // causal mask (j > i)
            if (j0 + 16 + nn > irow) x1 = -1e30f;
            float mx = fmaxf(x0, x1);
            for (int sh = 1; sh < 16; sh <<= 1) mx = fmaxf(mx, __shfl_xor(mx, sh, 32));
            float mn = fmaxf(mr[v], mx);
            float cr = __expf(mr[v] - mn);
            p0[v] = __expf(x0 - mn);
            p1[v] = __expf(x1 - mn);
            float rs = p0[v] + p1[v];
            for (int sh = 1; sh < 16; sh <<= 1) rs += __shfl_xor(rs, sh, 32);
            lr[v] = lr[v] * cr + rs;
            mr[v] = mn;
            o0[v] *= cr; o1[v] *= cr; o2[v] *= cr; o3[v] *= cr;
        }
        // P (C-layout) -> LDS -> A-fragment, then O += P x V
#pragma unroll
        for (int v = 0; v < 8; ++v) {
            int M = v + 8 * g;
            pb[M * 32 + nn]      = (_Float16)p0[v];
            pb[M * 32 + 16 + nn] = (_Float16)p1[v];
        }
        asm volatile("s_wait_dscnt 0" ::: "memory");
        h16 ap;
        {
            const _Float16* pp = pb + (size_t)nn * 32 + 8 * g;
            ((uint4*)&ap)[0] = *(const uint4*)(pp);
            ((uint4*)&ap)[1] = *(const uint4*)(pp + 16);
        }
        o0 = wmma16(ap, ldgB(Vp, QL, 0,  j0), o0);
        o1 = wmma16(ap, ldgB(Vp, QL, 16, j0), o1);
        o2 = wmma16(ap, ldgB(Vp, QL, 32, j0), o2);
        o3 = wmma16(ap, ldgB(Vp, QL, 48, j0), o3);
    }
    // normalize and emit attn_vec f16 [token][1024]
#pragma unroll
    for (int v = 0; v < 8; ++v) {
        int M = v + 8 * g;
        int tok = (i0 + M) * BS + (bh >> 4);
        float inv = 1.0f / lr[v];
        size_t base = (size_t)tok * DM + head * DH;
        AV[base + nn]      = (_Float16)(o0[v] * inv);
        AV[base + 16 + nn] = (_Float16)(o1[v] * inv);
        AV[base + 32 + nn] = (_Float16)(o2[v] * inv);
        AV[base + 48 + nn] = (_Float16)(o3[v] * inv);
    }
}

// ---------------- output projection + residual: x = w + AV @ Wo^T ----------------
__global__ __launch_bounds__(128) void k_oproj(const _Float16* __restrict__ AV,
                                               const _Float16* __restrict__ Wo,
                                               const float* __restrict__ w,
                                               float* __restrict__ XB) {
    int wid = blockIdx.x * 4 + (threadIdx.x >> 5);
    int m0 = (wid & 255) * 16, n0 = (wid >> 8) * 16;
    f8 acc = {0,0,0,0,0,0,0,0};
    for (int kk = 0; kk < DM; kk += 32)
        acc = wmma16(ldgA(AV, DM, m0, kk), ldgB(Wo, DM, n0, kk), acc);
    int lane = threadIdx.x & 31, g = lane >> 4, nn = lane & 15;
#pragma unroll
    for (int v = 0; v < 8; ++v) {
        size_t idx = (size_t)(m0 + v + 8 * g) * DM + n0 + nn;
        XB[idx] = acc[v] + w[idx];
    }
}

// ---------------- layernorm over last dim ----------------
__global__ __launch_bounds__(256) void k_ln(const float* __restrict__ XB,
                                            const float* __restrict__ gamma,
                                            const float* __restrict__ beta,
                                            float* __restrict__ out) {
    __shared__ float r1[256], r2[256];
    int t = blockIdx.x;
    const float* x = XB + (size_t)t * DM;
    float s = 0.f, s2 = 0.f;
    for (int c = threadIdx.x; c < DM; c += 256) { float v = x[c]; s += v; s2 += v * v; }
    r1[threadIdx.x] = s; r2[threadIdx.x] = s2; __syncthreads();
    for (int st = 128; st > 0; st >>= 1) {
        if (threadIdx.x < st) { r1[threadIdx.x] += r1[threadIdx.x + st]; r2[threadIdx.x] += r2[threadIdx.x + st]; }
        __syncthreads();
    }
    float mu = r1[0] * (1.0f / DM);
    float var = r2[0] * (1.0f / DM) - mu * mu;
    float rin = rsqrtf(var + 1e-5f);
    for (int c = threadIdx.x; c < DM; c += 256)
        out[(size_t)t * DM + c] = (x[c] - mu) * rin * gamma[c] + beta[c];
}

// ---------------- host: carve workspace, launch pipeline ----------------
extern "C" void kernel_launch(void* const* d_in, const int* in_sizes, int n_in,
                              void* d_out, int out_size, void* d_ws, size_t ws_size,
                              hipStream_t stream) {
    const float* w    = (const float*)d_in[0];   // [2048,2,1024]  (= [token][1024])
    const float* r    = (const float*)d_in[1];   // [2048,1024]
    const float* rwb  = (const float*)d_in[2];   // [16,64]
    const float* rrb  = (const float*)d_in[3];
    const float* Wqkv = (const float*)d_in[4];   // [3072,1024]
    const float* Wr   = (const float*)d_in[5];   // [1024,1024]
    const float* Wo   = (const float*)d_in[6];   // [1024,1024]
    const float* gam  = (const float*)d_in[7];
    const float* bet  = (const float*)d_in[8];
    float* out = (float*)d_out;

    char* p = (char*)d_ws;
    size_t off = 0;
    auto carve = [&](size_t bytes) { char* q = p + off; off = (off + bytes + 255) & ~size_t(255); return q; };
    _Float16* wf16   = (_Float16*)carve((size_t)TOK * DM * 2);       // 8 MB
    _Float16* wqkv16 = (_Float16*)carve((size_t)3 * DM * DM * 2);    // 6 MB
    _Float16* r16    = (_Float16*)carve((size_t)QL * DM * 2);        // 4 MB
    _Float16* wr16   = (_Float16*)carve((size_t)DM * DM * 2);        // 2 MB
    _Float16* wo16   = (_Float16*)carve((size_t)DM * DM * 2);        // 2 MB
    _Float16* Qw     = (_Float16*)carve((size_t)BS * NH * QL * DH * 2); // 8 MB
    _Float16* Qr     = (_Float16*)carve((size_t)BS * NH * QL * DH * 2);
    _Float16* Kf     = (_Float16*)carve((size_t)BS * NH * QL * DH * 2);
    _Float16* Vt     = (_Float16*)carve((size_t)BS * NH * DH * QL * 2);
    _Float16* RK     = (_Float16*)carve((size_t)NH * QL * DH * 2);   // 4 MB
    _Float16* AV     = (_Float16*)carve((size_t)TOK * DM * 2);       // 8 MB
    float*    XB     = (float*)   carve((size_t)TOK * DM * 4);       // 16 MB
    (void)ws_size; (void)in_sizes; (void)n_in; (void)out_size;

    k_cvt<<<4096, 256, 0, stream>>>(w,    wf16,   TOK * DM);
    k_cvt<<<3072, 256, 0, stream>>>(Wqkv, wqkv16, 3 * DM * DM);
    k_cvt<<<2048, 256, 0, stream>>>(r,    r16,    QL * DM);
    k_cvt<<<1024, 256, 0, stream>>>(Wr,   wr16,   DM * DM);
    k_cvt<<<1024, 256, 0, stream>>>(Wo,   wo16,   DM * DM);

    k_qkv  <<<12288, 128, 0, stream>>>(wf16, wqkv16, rwb, rrb, Qw, Qr, Kf, Vt); // 256x192 tiles
    k_rproj<<< 2048, 128, 0, stream>>>(r16, wr16, RK);                          // 128x64 tiles
    k_attn <<< 1024, 128, 0, stream>>>(Qw, Qr, Kf, Vt, RK, AV);                 // 4096 waves
    k_oproj<<< 4096, 128, 0, stream>>>(AV, wo16, w, XB);                        // 256x64 tiles
    k_ln   <<< 4096, 256, 0, stream>>>(XB, gam, bet, out);
}
